// xPaiNN_89713276879264
// MI455X (gfx1250) — matrix operand order repaired and checked by reading, not compiled, not run on hardware
//
#include <hip/hip_runtime.h>
#include <math.h>

#define BATCH 16
#define KAT   32
#define DA    128
#define DSZ   256
#define BM    26
#define BDIM  (KAT*BM)            // 832
#define NOCC  128
#define TDS   (3*DSZ)             // 768
#define BM2   (BM*BM)             // 676
#define NPAIR (KAT*(KAT-1)/2)     // 496
#define NPJ   (BDIM/2)            // 416
#define SCALEF 1e-4f
#define A0C    0.5291772f
#define JACOBI_SWEEPS 8

typedef __attribute__((ext_vector_type(16))) _Float16 v16h;
typedef __attribute__((ext_vector_type(8)))  _Float16 v8h;
typedef __attribute__((ext_vector_type(8)))  float    v8f;

// ---------------------------------------------------------------- prep
__global__ void f32_to_f16_kernel(const float* __restrict__ src,
                                  _Float16* __restrict__ dst, int n) {
  int i = blockIdx.x * blockDim.x + threadIdx.x;
  if (i < n) dst[i] = (_Float16)src[i];
}

// src: rows x cols (f32, row-major) -> dst: cols x rows (f16)
__global__ void transpose_f16_kernel(const float* __restrict__ src,
                                     _Float16* __restrict__ dst,
                                     int rows, int cols) {
  int i = blockIdx.x * blockDim.x + threadIdx.x;
  if (i < rows * cols) {
    int r = i / cols, c = i % cols;
    dst[(size_t)c * rows + r] = (_Float16)src[i];
  }
}

// ---------------------------------------------------------------- WMMA GEMM
// C[M,N] = A[M,Kd] (f16, row-major) * Bt[N,Kd]^T (f16, W pre-transposed)
//          + bias_mult * bias[N]
// one 16x16 tile per wave, 4 waves/block spanning 64 output columns.
__global__ __launch_bounds__(128)
void gemm_wmma_f16(const _Float16* __restrict__ A,
                   const _Float16* __restrict__ Bt,
                   const float* __restrict__ bias, float bias_mult,
                   float* __restrict__ Cf, _Float16* __restrict__ Ch,
                   int M, int N, int Kd) {
  const int lane = threadIdx.x & 31;
  const int wave = threadIdx.x >> 5;
  const int m0 = blockIdx.y * 16;
  const int n0 = blockIdx.x * 64 + wave * 16;
  if (m0 >= M) return;                       // wave-uniform

  const int half = lane >> 4;                // 0 or 1
  const int l16  = lane & 15;

  const int arow = m0 + l16;                 // M multiple of 16 in all uses
  int bcol = n0 + l16; if (bcol >= N) bcol = N - 1;   // clamp OOB cols
  const _Float16* aptr = A  + (size_t)arow * Kd;
  const _Float16* bptr = Bt + (size_t)bcol * Kd;

  v8f c = {};
  for (int k0 = 0; k0 < Kd; k0 += 32) {
    __builtin_prefetch(aptr + k0 + 64, 0, 1);
    __builtin_prefetch(bptr + k0 + 64, 0, 1);
    union { v16h v; v8h h[2]; } ua, ub;
    // A fragment (16x32 f16): lane half selects K groups {0..7,16..23} / {8..15,24..31}
    ua.h[0] = *(const v8h*)(aptr + k0 + 8 * half);
    ua.h[1] = *(const v8h*)(aptr + k0 + 16 + 8 * half);
    // B fragment (32x16): lane = column, lane half selects K groups {0..7,16..23} / {8..15,24..31}
    ub.h[0] = *(const v8h*)(bptr + k0 + 8 * half);
    ub.h[1] = *(const v8h*)(bptr + k0 + 16 + 8 * half);
    c = __builtin_amdgcn_wmma_f32_16x16x32_f16(false, ua.v, false, ub.v,
                                               (short)0, c, false, false);
  }

  const int col = n0 + l16;
  if (col < N) {
    const float bv = bias ? bias_mult * bias[col] : 0.f;
    for (int r = 0; r < 8; ++r) {
      const int row = m0 + r + 8 * half;
      const float v = c[r] + bv;
      if (Cf) Cf[(size_t)row * N + col] = v;
      if (Ch) Ch[(size_t)row * N + col] = (_Float16)v;
    }
  }
}

// ---------------------------------------------------------------- P builders
__global__ void sumx_kernel(const _Float16* __restrict__ X,
                            _Float16* __restrict__ SXh) {
  int n = blockIdx.y;
  int j = blockIdx.x * blockDim.x + threadIdx.x;
  if (j < TDS) {
    float s = 0.f;
    const _Float16* base = X + (size_t)n * KAT * TDS + j;
    for (int k = 0; k < KAT; ++k) s += (float)base[(size_t)k * TDS];
    SXh[(size_t)n * TDS + j] = (_Float16)s;
  }
}

__global__ void build_pon_kernel(const _Float16* __restrict__ X,
                                 const _Float16* __restrict__ SXh,
                                 _Float16* __restrict__ P) {
  size_t i = (size_t)blockIdx.x * blockDim.x + threadIdx.x;
  if (i < (size_t)BATCH * KAT * TDS) {
    int j = (int)(i % TDS);
    int m = (int)(i / TDS);
    int n = m / KAT;
    P[i] = (_Float16)((float)X[i] * (float)SXh[(size_t)n * TDS + j]);
  }
}

__global__ void build_poff_kernel(const _Float16* __restrict__ X,
                                  _Float16* __restrict__ P) {
  size_t i = (size_t)blockIdx.x * blockDim.x + threadIdx.x;
  if (i < (size_t)BATCH * NPAIR * TDS) {
    int j = (int)(i % TDS);
    int m = (int)(i / TDS);
    int n = m / NPAIR;
    int p = m % NPAIR;
    int k = 1; while ((k * (k + 1)) / 2 <= p) ++k;   // k>l, p = k(k-1)/2 + l
    int l = p - (k * (k - 1)) / 2;
    const _Float16* Xn = X + (size_t)n * KAT * TDS;
    P[i] = (_Float16)((float)Xn[(size_t)k * TDS + j] *
                      (float)Xn[(size_t)l * TDS + j]);
  }
}

// ---------------------------------------------------------------- F assembly
__global__ void assemble_F_kernel(const float* __restrict__ ON,
                                  const float* __restrict__ OFF,
                                  const float* __restrict__ h_new,
                                  const float* __restrict__ F_offset,
                                  float* __restrict__ Fout,
                                  float* __restrict__ Aw) {
  size_t idx = (size_t)blockIdx.x * blockDim.x + threadIdx.x;
  const size_t total = (size_t)BATCH * BDIM * BDIM;
  if (idx >= total) return;
  int n   = (int)(idx / ((size_t)BDIM * BDIM));
  int rem = (int)(idx % ((size_t)BDIM * BDIM));
  int r = rem / BDIM, cc = rem % BDIM;
  int k = r / BM, a = r % BM;
  int l = cc / BM, b = cc % BM;
  float v;
  if (k == l) {
    const float* on = ON + (size_t)(n * KAT + k) * BM2;
    v = 0.5f * (on[a * BM + b] + on[b * BM + a]);
  } else if (k > l) {
    int p = (k * (k - 1)) / 2 + l;
    v = OFF[(size_t)(n * NPAIR + p) * BM2 + a * BM + b];
  } else {
    int p = (l * (l - 1)) / 2 + k;
    v = OFF[(size_t)(n * NPAIR + p) * BM2 + b * BM + a];
  }
  float h = h_new[idx];
  float f = ((fabsf(h) > 1e-7f) ? v * SCALEF : 0.f) + F_offset[rem];
  Fout[idx] = f;
  Aw[idx]   = f;
}

// ---------------------------------------------------------------- Jacobi eigh
// One workgroup (32 wave32) per matrix. Parallel two-sided Jacobi with
// round-robin pairing: 416 disjoint rotations per round, 831 rounds/sweep.
__global__ __launch_bounds__(1024)
void jacobi_kernel(float* __restrict__ Aw, float* __restrict__ U) {
  __shared__ int   perm[BDIM];
  __shared__ int   perm2[BDIM];
  __shared__ float cs[NPJ], sn[NPJ];
  __shared__ int   pp[NPJ], qq[NPJ];
  const int n   = blockIdx.x;
  const int tid = threadIdx.x;
  float* A = Aw + (size_t)n * BDIM * BDIM;
  float* V = U  + (size_t)n * BDIM * BDIM;

  for (size_t i = tid; i < (size_t)BDIM * BDIM; i += blockDim.x) {
    int rr = (int)(i / BDIM), ccc = (int)(i % BDIM);
    V[i] = (rr == ccc) ? 1.f : 0.f;
  }
  for (int i = tid; i < BDIM; i += blockDim.x) perm[i] = i;
  __syncthreads();

  for (int sw = 0; sw < JACOBI_SWEEPS; ++sw)
    for (int rd = 0; rd < BDIM - 1; ++rd) {
      if (tid < NPJ) {
        int p = perm[tid], q = perm[BDIM - 1 - tid];
        if (p > q) { int t = p; p = q; q = t; }
        float app = A[(size_t)p * BDIM + p];
        float aqq = A[(size_t)q * BDIM + q];
        float apq = A[(size_t)p * BDIM + q];
        float c = 1.f, s = 0.f;
        if (fabsf(apq) > 1e-12f) {
          float th = 0.5f * (aqq - app) / apq;
          float t  = copysignf(1.f, th) / (fabsf(th) + sqrtf(1.f + th * th));
          c = 1.f / sqrtf(1.f + t * t);
          s = t * c;
        }
        cs[tid] = c; sn[tid] = s; pp[tid] = p; qq[tid] = q;
      }
      __syncthreads();
      // A <- A * J  (columns p,q; pairs disjoint)
      for (int t = tid; t < NPJ * BDIM; t += blockDim.x) {
        int j = t / BDIM, i = t % BDIM;
        int p = pp[j], q = qq[j]; float c = cs[j], s = sn[j];
        float ap = A[(size_t)i * BDIM + p], aq = A[(size_t)i * BDIM + q];
        A[(size_t)i * BDIM + p] = c * ap - s * aq;
        A[(size_t)i * BDIM + q] = s * ap + c * aq;
      }
      __syncthreads();
      // A <- J^T * A (rows p,q)
      for (int t = tid; t < NPJ * BDIM; t += blockDim.x) {
        int j = t / BDIM, i = t % BDIM;
        int p = pp[j], q = qq[j]; float c = cs[j], s = sn[j];
        float ap = A[(size_t)p * BDIM + i], aq = A[(size_t)q * BDIM + i];
        A[(size_t)p * BDIM + i] = c * ap - s * aq;
        A[(size_t)q * BDIM + i] = s * ap + c * aq;
      }
      __syncthreads();
      // V <- V * J
      for (int t = tid; t < NPJ * BDIM; t += blockDim.x) {
        int j = t / BDIM, i = t % BDIM;
        int p = pp[j], q = qq[j]; float c = cs[j], s = sn[j];
        float vp = V[(size_t)i * BDIM + p], vq = V[(size_t)i * BDIM + q];
        V[(size_t)i * BDIM + p] = c * vp - s * vq;
        V[(size_t)i * BDIM + q] = s * vp + c * vq;
      }
      __syncthreads();
      // advance tournament: fix perm[0], rotate the rest by one
      for (int i = tid; i < BDIM; i += blockDim.x) {
        int src = (i == 0) ? perm[0] : (i == 1 ? perm[BDIM - 1] : perm[i - 1]);
        perm2[i] = src;
      }
      __syncthreads();
      for (int i = tid; i < BDIM; i += blockDim.x) perm[i] = perm2[i];
      __syncthreads();
    }
}

// ---------------------------------------------------------------- sort eigs
__global__ __launch_bounds__(1024)
void sort_eigs_kernel(const float* __restrict__ Aw,
                      float* __restrict__ e_out, int* __restrict__ idx_out) {
  __shared__ float val[1024];
  __shared__ int   idx[1024];
  const int n = blockIdx.x, tid = threadIdx.x;
  const float* A = Aw + (size_t)n * BDIM * BDIM;
  val[tid] = (tid < BDIM) ? A[(size_t)tid * BDIM + tid] : 3.4e38f;
  idx[tid] = tid;
  __syncthreads();
  for (int ksz = 2; ksz <= 1024; ksz <<= 1)
    for (int jsz = ksz >> 1; jsz > 0; jsz >>= 1) {
      int ixj = tid ^ jsz;
      if (ixj > tid) {
        bool up = ((tid & ksz) == 0);
        bool sw = up ? (val[tid] > val[ixj]) : (val[tid] < val[ixj]);
        if (sw) {
          float tv = val[tid]; val[tid] = val[ixj]; val[ixj] = tv;
          int   ti = idx[tid]; idx[tid] = idx[ixj]; idx[ixj] = ti;
        }
      }
      __syncthreads();
    }
  if (tid < BDIM) {
    e_out[(size_t)n * BDIM + tid]   = val[tid];
    idx_out[(size_t)n * BDIM + tid] = idx[tid];
  }
}

// ---------------------------------------------------------------- u^T H u
__global__ __launch_bounds__(256)
void trace_kernel(const float* __restrict__ U, const int* __restrict__ idxs,
                  const float* __restrict__ h_new, float* __restrict__ tr_out) {
  __shared__ float uv[BDIM];
  __shared__ float red[256];
  const int n = blockIdx.y, i = blockIdx.x, tid = threadIdx.x;
  const int col = idxs[(size_t)n * BDIM + i];
  const float* V = U     + (size_t)n * BDIM * BDIM;
  const float* H = h_new + (size_t)n * BDIM * BDIM;
  for (int r = tid; r < BDIM; r += 256) uv[r] = V[(size_t)r * BDIM + col];
  __syncthreads();
  float acc = 0.f;
  for (int r = tid; r < BDIM; r += 256) {
    const float* Hr = H + (size_t)r * BDIM;
    float inner = 0.f;
    for (int c = 0; c < BDIM; ++c) inner += Hr[c] * uv[c];
    acc += uv[r] * inner;
  }
  red[tid] = acc; __syncthreads();
  for (int s = 128; s > 0; s >>= 1) {
    if (tid < s) red[tid] += red[tid + s];
    __syncthreads();
  }
  if (tid == 0) tr_out[(size_t)n * NOCC + i] = red[0];
}

// ---------------------------------------------------------------- energy
__global__ __launch_bounds__(256)
void energy_kernel(const float* __restrict__ e_sorted,
                   const float* __restrict__ tr,
                   const float* __restrict__ positions,
                   const int* __restrict__ numbers,
                   float* __restrict__ E_out) {
  __shared__ float red[256];
  const int n = blockIdx.x, tid = threadIdx.x;
  float acc = 0.f;
  for (int i = tid; i < NOCC; i += 256)
    acc += e_sorted[(size_t)n * BDIM + i] + tr[(size_t)n * NOCC + i];
  for (int p = tid; p < NPAIR; p += 256) {
    int k = 1; while ((k * (k + 1)) / 2 <= p) ++k;
    int l = p - (k * (k - 1)) / 2;
    const float* pk = positions + ((size_t)n * KAT + k) * 3;
    const float* pl = positions + ((size_t)n * KAT + l) * 3;
    float dx = pk[0] - pl[0], dy = pk[1] - pl[1], dz = pk[2] - pl[2];
    float d = sqrtf(dx * dx + dy * dy + dz * dz);
    float Z = (float)(numbers[k] * numbers[l]);
    acc += A0C * Z / d;
  }
  red[tid] = acc; __syncthreads();
  for (int s = 128; s > 0; s >>= 1) {
    if (tid < s) red[tid] += red[tid + s];
    __syncthreads();
  }
  if (tid == 0) E_out[n] = red[0];
}

// ---------------------------------------------------------------- launch
extern "C" void kernel_launch(void* const* d_in, const int* in_sizes, int n_in,
                              void* d_out, int out_size, void* d_ws, size_t ws_size,
                              hipStream_t stream) {
  const float* mu         = (const float*)d_in[0];
  const float* h_new      = (const float*)d_in[1];
  const float* positions  = (const float*)d_in[2];
  const float* F_offset   = (const float*)d_in[3];
  const float* transfer_W = (const float*)d_in[4];
  const float* transfer_b = (const float*)d_in[5];
  const float* WF_on_W    = (const float*)d_in[6];
  const float* WF_on_b    = (const float*)d_in[7];
  const float* WF_off_W   = (const float*)d_in[8];
  const float* WF_off_b   = (const float*)d_in[9];
  const int*   numbers    = (const int*)d_in[10];

  // workspace carve-out
  char* ws = (char*)d_ws; size_t off = 0;
  auto carve = [&](size_t bytes) {
    char* p = ws + off;
    off += (bytes + 255) & ~(size_t)255;
    return p;
  };
  const int MT = BATCH * KAT * 3;                 // 1536 transfer rows
  _Float16* muh    = (_Float16*)carve((size_t)MT * DA * 2);
  _Float16* Wt_tr  = (_Float16*)carve((size_t)DSZ * DA * 2);
  _Float16* Wt_on  = (_Float16*)carve((size_t)BM2 * TDS * 2);
  _Float16* Wt_off = (_Float16*)carve((size_t)BM2 * TDS * 2);
  _Float16* X      = (_Float16*)carve((size_t)BATCH * KAT * TDS * 2);
  _Float16* SXh    = (_Float16*)carve((size_t)BATCH * TDS * 2);
  _Float16* P_on   = (_Float16*)carve((size_t)BATCH * KAT * TDS * 2);
  _Float16* P_off  = (_Float16*)carve((size_t)BATCH * NPAIR * TDS * 2);
  float*    ON     = (float*)carve((size_t)BATCH * KAT * BM2 * 4);
  float*    OFF    = (float*)carve((size_t)BATCH * NPAIR * BM2 * 4);
  float*    Aw     = (float*)carve((size_t)BATCH * BDIM * BDIM * 4);
  float*    U      = (float*)carve((size_t)BATCH * BDIM * BDIM * 4);
  int*      eidx   = (int*)carve((size_t)BATCH * BDIM * 4);
  float*    trbuf  = (float*)carve((size_t)BATCH * NOCC * 4);
  (void)ws_size; (void)in_sizes; (void)n_in; (void)out_size;

  float* Fout = (float*)d_out;
  float* eout = Fout + (size_t)BATCH * BDIM * BDIM;
  float* Eout = eout + (size_t)BATCH * BDIM;

  // 1) conversions / transposes
  {
    int n = MT * DA;
    f32_to_f16_kernel<<<(n + 255) / 256, 256, 0, stream>>>(mu, muh, n);
    transpose_f16_kernel<<<(DA * DSZ + 255) / 256, 256, 0, stream>>>(transfer_W, Wt_tr, DA, DSZ);
    transpose_f16_kernel<<<(TDS * BM2 + 255) / 256, 256, 0, stream>>>(WF_on_W, Wt_on, TDS, BM2);
    transpose_f16_kernel<<<(TDS * BM2 + 255) / 256, 256, 0, stream>>>(WF_off_W, Wt_off, TDS, BM2);
  }
  // 2) x = mu @ transfer_W + b  -> X (f16), row (n,k,d) x 256 == (n,k,768) layout
  gemm_wmma_f16<<<dim3((DSZ + 63) / 64, MT / 16), 128, 0, stream>>>(
      muh, Wt_tr, transfer_b, 1.f, nullptr, X, MT, DSZ, DA);
  // 3) SX and P matrices
  sumx_kernel<<<dim3((TDS + 255) / 256, BATCH), 256, 0, stream>>>(X, SXh);
  {
    size_t n1 = (size_t)BATCH * KAT * TDS;
    build_pon_kernel<<<(unsigned)((n1 + 255) / 256), 256, 0, stream>>>(X, SXh, P_on);
    size_t n2 = (size_t)BATCH * NPAIR * TDS;
    build_poff_kernel<<<(unsigned)((n2 + 255) / 256), 256, 0, stream>>>(X, P_off);
  }
  // 4) on / off GEMMs (WMMA f16 -> f32)
  gemm_wmma_f16<<<dim3((BM2 + 63) / 64, (BATCH * KAT) / 16), 128, 0, stream>>>(
      P_on, Wt_on, WF_on_b, (float)KAT, ON, nullptr, BATCH * KAT, BM2, TDS);
  gemm_wmma_f16<<<dim3((BM2 + 63) / 64, (BATCH * NPAIR) / 16), 128, 0, stream>>>(
      P_off, Wt_off, WF_off_b, 1.f, OFF, nullptr, BATCH * NPAIR, BM2, TDS);
  // 5) assemble F (writes d_out F and Jacobi working copy)
  {
    size_t total = (size_t)BATCH * BDIM * BDIM;
    assemble_F_kernel<<<(unsigned)((total + 255) / 256), 256, 0, stream>>>(
        ON, OFF, h_new, F_offset, Fout, Aw);
  }
  // 6) eigendecomposition
  jacobi_kernel<<<BATCH, 1024, 0, stream>>>(Aw, U);
  sort_eigs_kernel<<<BATCH, 1024, 0, stream>>>(Aw, eout, eidx);
  // 7) occupied-subspace energy + nuclear repulsion
  trace_kernel<<<dim3(NOCC, BATCH), 256, 0, stream>>>(U, eidx, h_new, trbuf);
  energy_kernel<<<BATCH, 256, 0, stream>>>(eout, trbuf, positions, numbers, Eout);
}
